// SlidingWindowAttention_66486093742227
// MI455X (gfx1250) — matrix operand order repaired and checked by reading, not compiled, not run on hardware
//
#include <hip/hip_runtime.h>
#include <hip/hip_bf16.h>
#include <cstdint>
#include <cstddef>

// ---------------------------------------------------------------------------
// MI455X (gfx1250) attention block.
// Roofline: ~77 GFLOP, ~100 MB HBM traffic -> ~5us at 23.3 TB/s if all GEMM
// work runs on v_wmma_f32_16x16x32_bf16 (fp32 WMMA is 16x16x4 only -> 8x less
// K/instr). So: convert to bf16 once, do every matmul-shaped op via WMMA with
// fp32 accumulation, keep softmax/RMS/RoPE in fp32 VALU.
// ---------------------------------------------------------------------------

typedef __bf16 bf16;
typedef __attribute__((ext_vector_type(16))) __bf16 v16bf;
typedef __attribute__((ext_vector_type(8)))  __bf16 v8bf;
typedef __attribute__((ext_vector_type(8)))  float  v8f;

#define WMMA_BF16(a, b, c) \
  __builtin_amdgcn_wmma_f32_16x16x32_bf16(false, (a), false, (b), (short)0, (c), false, false)

constexpr int BB   = 2;
constexpr int S    = 2048;
constexpr int H    = 2048;
constexpr int NH   = 16;
constexpr int NKV  = 4;
constexpr int HD   = 64;
constexpr int BS   = BB * S;                 // 4096 tokens
constexpr int DQ   = NH * HD;                // 1024
constexpr int DKV  = NKV * HD;               // 256
constexpr int DQKV = DQ + 2 * DKV;           // 1536

// Build a 16x bf16 fragment from two 16-byte halves (ISA A/B VGPR layouts:
// element 2v,2v+1 live in VGPR v -> plain increasing-K order in the vector).
__device__ __forceinline__ v16bf load_frag(const bf16* lo, const bf16* hi) {
  v8bf l = *(const v8bf*)lo;
  v8bf h = *(const v8bf*)hi;
  v16bf r;
#pragma unroll
  for (int i = 0; i < 8; ++i) { r[i] = l[i]; r[i + 8] = h[i]; }
  return r;
}

// ---------------------------------------------------------------------------
// Kernel: fp32 -> bf16 copy (x)
// ---------------------------------------------------------------------------
__global__ void cvt_f32_bf16_kernel(const float* __restrict__ src,
                                    bf16* __restrict__ dst, int n) {
  int i = blockIdx.x * blockDim.x + threadIdx.x;
  int stride = gridDim.x * blockDim.x;
  for (; i < n; i += stride) dst[i] = (bf16)src[i];
}

// ---------------------------------------------------------------------------
// Kernel: transpose + convert weights: dst[n*dstStride + k] = src[k*N + n]
// Gives B^T row-major so WMMA B fragments are contiguous 16-element loads.
// ---------------------------------------------------------------------------
__global__ void transpose_cvt_kernel(const float* __restrict__ src,
                                     bf16* __restrict__ dst,
                                     int K, int N, int dstStride) {
  int i = blockIdx.x * blockDim.x + threadIdx.x;
  int stride = gridDim.x * blockDim.x;
  int total = K * N;
  for (; i < total; i += stride) {
    int kk = i / N;
    int nn = i - kk * N;
    dst[(size_t)nn * dstStride + kk] = (bf16)src[i];
  }
}

// ---------------------------------------------------------------------------
// Generic bf16 WMMA GEMM:  C[M,N] (f32) = A[M,K] (bf16, row-major)
//                                        x Bt[N,K] (bf16, = B^T row-major)
// 128 threads = 4 waves; each wave owns a 16x64 tile (4 accumulators reusing
// one A fragment per 32-K step). All operands L2-resident (192MB L2).
// ---------------------------------------------------------------------------
__global__ __launch_bounds__(128)
void gemm_bf16_kernel(const bf16* __restrict__ A, const bf16* __restrict__ Bt,
                      float* __restrict__ C, int M, int N, int K) {
  const int lane = threadIdx.x & 31;
  const int wave = threadIdx.x >> 5;
  const int col  = lane & 15;
  const int half = (lane >> 4) & 1;      // wave32 halves
  const int mrow = blockIdx.y * 64 + wave * 16;
  const int nbase = blockIdx.x * 64;

  const v8f z = {0.f, 0.f, 0.f, 0.f, 0.f, 0.f, 0.f, 0.f};
  v8f acc[4];
#pragma unroll
  for (int t = 0; t < 4; ++t) acc[t] = z;

  // A fragment: row = mrow+col, K halves at +0/+16 (lanes<16) or +8/+24.
  const bf16* pa = A + (size_t)(mrow + col) * K + half * 8;

  for (int kk = 0; kk < K; kk += 32) {
    v16bf a = load_frag(pa, pa + 16);
    pa += 32;
    __builtin_prefetch((const void*)(pa + 32), 0, 0);   // global_prefetch_b8
#pragma unroll
    for (int t = 0; t < 4; ++t) {
      const bf16* pb = Bt + (size_t)(nbase + t * 16 + col) * K + kk + half * 8;
      v16bf b = load_frag(pb, pb + 16);
      acc[t] = WMMA_BF16(a, b, acc[t]);
    }
  }

  const int hi8 = half * 8;   // C/D layout: VGPR r <-> M = r + 8*(lane>=16)
#pragma unroll
  for (int t = 0; t < 4; ++t)
#pragma unroll
    for (int r = 0; r < 8; ++r)
      C[(size_t)(mrow + r + hi8) * N + nbase + t * 16 + col] = acc[t][r];
}

// ---------------------------------------------------------------------------
// RMS-norm + RoPE epilogue on the fp32 QKV GEMM output; emits bf16 operands
// laid out for the attention WMMAs: q/k row-major [b,h,s,d], v transposed
// [b,h,d,s] so the P@V B-fragment is 16 contiguous keys.
// One 64-thread block per (token, logical head).
// ---------------------------------------------------------------------------
__global__ __launch_bounds__(64)
void norm_rope_kernel(const float* __restrict__ qkv,
                      const float* __restrict__ cosb, const float* __restrict__ sinb,
                      const float* __restrict__ qw,   const float* __restrict__ kw,
                      bf16* __restrict__ qo, bf16* __restrict__ ko,
                      bf16* __restrict__ vto) {
  __shared__ float sh[HD];
  const int token = blockIdx.x;
  const int hid   = blockIdx.y;           // 0..15 q, 16..19 k, 20..23 v
  const int d     = threadIdx.x;
  const int b     = token / S;
  const int s     = token - b * S;
  const float* base = qkv + (size_t)token * DQKV;

  if (hid < NH + NKV) {                   // q or k: rms-norm + rope
    const bool isq = hid < NH;
    const int  hh  = isq ? hid : hid - NH;
    const float x  = isq ? base[hh * HD + d] : base[DQ + hh * HD + d];
    sh[d] = x * x;
    __syncthreads();
    float var = 0.f;
#pragma unroll 8
    for (int i = 0; i < HD; ++i) var += sh[i];
    var *= (1.f / HD);
    __syncthreads();
    const float w  = isq ? qw[d] : kw[d];
    const float xn = x * rsqrtf(var + 1e-6f) * w;
    sh[d] = xn;
    __syncthreads();
    const float rot = (d < 32) ? -sh[d + 32] : sh[d - 32];
    const float y   = xn * cosb[s * HD + d] + rot * sinb[s * HD + d];
    if (isq)
      qo[((size_t)(b * NH + hh) * S + s) * HD + d] = (bf16)y;
    else
      ko[((size_t)(b * NKV + hh) * S + s) * HD + d] = (bf16)y;
  } else {                                // v: transpose-store only
    const int hh = hid - NH - NKV;
    const float x = base[DQ + DKV + hh * HD + d];
    vto[((size_t)(b * NKV + hh) * HD + d) * S + s] = (bf16)x;
  }
}

// ---------------------------------------------------------------------------
// Causal flash attention, one wave per 16-query tile per head.
// Per 32-key block: 4 score WMMAs (2 N-tiles x HD/32 K-steps), online softmax
// (row reductions with shfl_xor across the 16-lane N group of the C layout),
// P staged through LDS to re-gather the A-fragment layout, 4 P@V WMMAs.
// ---------------------------------------------------------------------------
__global__ __launch_bounds__(32)
void attn_kernel(const bf16* __restrict__ q, const bf16* __restrict__ k,
                 const bf16* __restrict__ vt, bf16* __restrict__ ao) {
  __shared__ bf16 plds[16 * 32];          // P tile, row-major [16 q][32 keys]

  const int lane = threadIdx.x;
  const int q0   = blockIdx.x * 16;
  const int h    = blockIdx.y;
  const int b    = blockIdx.z;
  const int hk   = h >> 2;                // GQA: 4 q-heads share one kv-head
  const int col  = lane & 15;
  const int half = (lane >> 4) & 1;
  const int hi8  = half * 8;
  const int ka   = half * 8;              // A-layout K offset per lane half

  const bf16* qb = q + (size_t)(b * NH + h) * S * HD + (size_t)(q0 + col) * HD;
  const v16bf a0 = load_frag(qb + ka,      qb + ka + 16);   // K 0..31 of HD
  const v16bf a1 = load_frag(qb + 32 + ka, qb + 32 + ka + 16); // K 32..63

  const v8f z = {0.f, 0.f, 0.f, 0.f, 0.f, 0.f, 0.f, 0.f};
  float m[8], l[8];
  v8f o[4];
#pragma unroll
  for (int r = 0; r < 8; ++r) { m[r] = -INFINITY; l[r] = 0.f; }
#pragma unroll
  for (int n = 0; n < 4; ++n) o[n] = z;

  const bf16* kbase  = k  + (size_t)(b * NKV + hk) * S * HD;
  const bf16* vtbase = vt + (size_t)(b * NKV + hk) * HD * S;
  const int kend = q0 + 16;

  for (int kb = 0; kb < kend; kb += 32) {
    // ---- scores: S = Q x K^T  (B[dk][key] = k[key][dk], contiguous in dk)
    v8f sc[2];
    sc[0] = z; sc[1] = z;
#pragma unroll
    for (int nt = 0; nt < 2; ++nt) {
      const int key = kb + nt * 16 + col;
      const bf16* kp = kbase + (size_t)key * HD + half * 8;
      v16bf b0 = load_frag(kp,      kp + 16);      // dk 0..31 half
      v16bf b1 = load_frag(kp + 32, kp + 48);      // dk 32..63 half
      sc[nt] = WMMA_BF16(a0, b0, sc[nt]);
      sc[nt] = WMMA_BF16(a1, b1, sc[nt]);
    }

    // ---- online softmax (per-row: M = q0 + r + hi8, row lives in 16 lanes)
    float alpha[8];
#pragma unroll
    for (int r = 0; r < 8; ++r) {
      const int Mrow = q0 + r + hi8;
      const int N0 = kb + col;
      float s0 = (N0      <= Mrow) ? sc[0][r] * 0.125f : -INFINITY;
      float s1 = (N0 + 16 <= Mrow) ? sc[1][r] * 0.125f : -INFINITY;
      float mx = fmaxf(s0, s1);
#pragma unroll
      for (int off = 1; off < 16; off <<= 1) mx = fmaxf(mx, __shfl_xor(mx, off, 32));
      const float mn = fmaxf(m[r], mx);
      const float al = __expf(m[r] - mn);
      const float p0 = __expf(s0 - mn);
      const float p1 = __expf(s1 - mn);
      float rs = p0 + p1;
#pragma unroll
      for (int off = 1; off < 16; off <<= 1) rs += __shfl_xor(rs, off, 32);
      l[r] = l[r] * al + rs;
      m[r] = mn;
      alpha[r] = al;
      plds[(r + hi8) * 32 + col]      = (bf16)p0;   // D-layout -> row-major LDS
      plds[(r + hi8) * 32 + col + 16] = (bf16)p1;
    }
    __syncthreads();
    // re-gather P as an A fragment (row = col lane, K halves at ka/ka+16)
    const v16bf pa = load_frag(&plds[col * 32 + ka], &plds[col * 32 + ka + 16]);
    __syncthreads();

    // ---- O = diag(alpha) O + P x V   (Vt[d][key]: contiguous 16 keys)
#pragma unroll
    for (int n = 0; n < 4; ++n) {
#pragma unroll
      for (int r = 0; r < 8; ++r) o[n][r] *= alpha[r];
      const bf16* vp = vtbase + (size_t)(n * 16 + col) * S + kb + half * 8;
      v16bf bv = load_frag(vp, vp + 16);
      o[n] = WMMA_BF16(pa, bv, o[n]);
    }
  }

  // ---- epilogue: normalize and store bf16 [token][h*64+d]
#pragma unroll
  for (int r = 0; r < 8; ++r) l[r] = 1.f / l[r];
#pragma unroll
  for (int n = 0; n < 4; ++n)
#pragma unroll
    for (int r = 0; r < 8; ++r) {
      const int row = q0 + r + hi8;
      ao[((size_t)(b * S + row)) * DQ + h * HD + n * 16 + col] =
          (bf16)(o[n][r] * l[r]);
    }
}

// ---------------------------------------------------------------------------
// Launch
// ---------------------------------------------------------------------------
extern "C" void kernel_launch(void* const* d_in, const int* in_sizes, int n_in,
                              void* d_out, int out_size, void* d_ws, size_t ws_size,
                              hipStream_t stream) {
  const float* x  = (const float*)d_in[0];
  const float* cs = (const float*)d_in[1];
  const float* sn = (const float*)d_in[2];
  const float* wq = (const float*)d_in[3];
  const float* wk = (const float*)d_in[4];
  const float* wv = (const float*)d_in[5];
  const float* wo = (const float*)d_in[6];
  const float* qw = (const float*)d_in[7];
  const float* kw = (const float*)d_in[8];
  float* out = (float*)d_out;

  char* ws = (char*)d_ws;
  size_t off = 0;
  bf16* xb     = (bf16*)(ws + off); off += (size_t)BS * H * sizeof(bf16);       // 16.8MB
  bf16* wqkv_t = (bf16*)(ws + off); off += (size_t)DQKV * H * sizeof(bf16);     //  6.3MB
  bf16* wo_t   = (bf16*)(ws + off); off += (size_t)H * DQ * sizeof(bf16);       //  4.2MB
  float* qkvf  = (float*)(ws + off); off += (size_t)BS * DQKV * sizeof(float);  // 25.2MB
  bf16* qbf    = (bf16*)(ws + off); off += (size_t)BB * NH * S * HD * sizeof(bf16);  // 8.4MB
  bf16* kbf    = (bf16*)(ws + off); off += (size_t)BB * NKV * S * HD * sizeof(bf16); // 2.1MB
  bf16* vtbf   = (bf16*)(ws + off); off += (size_t)BB * NKV * HD * S * sizeof(bf16); // 2.1MB
  bf16* aob    = (bf16*)(ws + off); off += (size_t)BS * DQ * sizeof(bf16);      //  8.4MB
  (void)off; (void)ws_size; (void)in_sizes; (void)n_in; (void)out_size;

  // 1) precision/layout prep (bf16; weights transposed for contiguous B frags)
  cvt_f32_bf16_kernel<<<4096, 256, 0, stream>>>(x, xb, BS * H);
  transpose_cvt_kernel<<<2048, 256, 0, stream>>>(wq, wqkv_t,                    H, DQ,  H);
  transpose_cvt_kernel<<<512,  256, 0, stream>>>(wk, wqkv_t + (size_t)DQ * H,   H, DKV, H);
  transpose_cvt_kernel<<<512,  256, 0, stream>>>(wv, wqkv_t + (size_t)(DQ + DKV) * H, H, DKV, H);
  transpose_cvt_kernel<<<2048, 256, 0, stream>>>(wo, wo_t, DQ, H, DQ);

  // 2) fused QKV projection (WMMA bf16 -> f32)
  gemm_bf16_kernel<<<dim3(DQKV / 64, BS / 64), 128, 0, stream>>>(
      xb, wqkv_t, qkvf, BS, DQKV, H);

  // 3) RMS-norm + RoPE, emit attention-layout bf16 operands
  norm_rope_kernel<<<dim3(BS, NH + 2 * NKV), 64, 0, stream>>>(
      qkvf, cs, sn, qw, kw, qbf, kbf, vtbf);

  // 4) causal flash attention (WMMA scores + WMMA P@V)
  attn_kernel<<<dim3(S / 16, NH, BB), 32, 0, stream>>>(qbf, kbf, vtbf, aob);

  // 5) output projection (WMMA bf16 -> f32) straight into d_out
  gemm_bf16_kernel<<<dim3(H / 64, BS / 64), 128, 0, stream>>>(
      aob, wo_t, out, BS, H, DQ);
}
